// MoELayer_4501125726439
// MI455X (gfx1250) — matrix operand order repaired and checked by reading, not compiled, not run on hardware
//
#include <hip/hip_runtime.h>
#include <hip/hip_bf16.h>

// ---------------- problem constants ----------------
#define NTOK 8192      // B*S
#define DDIM 512
#define FDIM 256
#define NEXP 64
#define NSH  2
#define TOPK 6
#define CAP  1536      // N*K*2.0/E

typedef unsigned short u16;
typedef unsigned int   u32;
typedef unsigned long long u64;

typedef __attribute__((ext_vector_type(16))) __bf16 bf16x16;
typedef __attribute__((ext_vector_type(8)))  float  f32x8;

__device__ __forceinline__ u16 f2bf(float f) {           // round-to-nearest-even
    u32 u = __float_as_uint(f);
    u32 r = u + 0x7fffu + ((u >> 16) & 1u);
    return (u16)(r >> 16);
}
__device__ __forceinline__ float bf2f(u16 h) {
    return __uint_as_float(((u32)h) << 16);
}

// async global->LDS 32B copy (two B128 ops; offset:16 advances BOTH addresses)
__device__ __forceinline__ void async_copy32(u32 lds_addr, const u16* gptr) {
    u64 ga = (u64)(size_t)gptr;
    asm volatile("global_load_async_to_lds_b128 %0, %1, off"
                 :: "v"(lds_addr), "v"(ga) : "memory");
    asm volatile("global_load_async_to_lds_b128 %0, %1, off offset:16"
                 :: "v"(lds_addr), "v"(ga) : "memory");
}
__device__ __forceinline__ void wait_async0() {
    asm volatile("s_wait_asynccnt 0" ::: "memory");
}
__device__ __forceinline__ u32 lds_off(const void* p) {   // flat LDS addr low 32b
    return (u32)(size_t)p;
}

// ---------------- 1) f32 -> bf16 convert (grid-stride, 4 elems/thread) ------
__global__ void k_cvt(const float* __restrict__ src, u16* __restrict__ dst, int n4) {
    int i = blockIdx.x * blockDim.x + threadIdx.x;
    int stride = gridDim.x * blockDim.x;
    for (; i < n4; i += stride) {
        float4 f = ((const float4*)src)[i];
        uint2 o;
        o.x = (u32)f2bf(f.x) | ((u32)f2bf(f.y) << 16);
        o.y = (u32)f2bf(f.z) | ((u32)f2bf(f.w) << 16);
        ((uint2*)dst)[i] = o;
    }
}

// ---------------- 2) router logits: one block per token, thread = expert ----
__global__ __launch_bounds__(64) void k_router(const float* __restrict__ x,
                                               const float* __restrict__ rw,
                                               const float* __restrict__ bias,
                                               float* __restrict__ logits) {
    __shared__ float xs[DDIM];
    int t = blockIdx.x;
    for (int i = threadIdx.x; i < DDIM; i += 64) xs[i] = x[(size_t)t * DDIM + i];
    __syncthreads();
    int e = threadIdx.x;
    const float* w = rw + (size_t)e * DDIM;
    float acc = 0.f;
    #pragma unroll 4
    for (int d = 0; d < DDIM; ++d) acc += xs[d] * w[d];
    logits[(size_t)t * NEXP + e] = acc + bias[e];
}

// ---------------- 3) softmax + top-6 + renorm: one thread per token ---------
__global__ void k_topk(const float* __restrict__ logits,
                       int* __restrict__ ti, float* __restrict__ tg) {
    int t = blockIdx.x * blockDim.x + threadIdx.x;
    if (t >= NTOK) return;
    float p[NEXP];
    const float* lp = logits + (size_t)t * NEXP;
    float m = -3.4e38f;
    for (int e = 0; e < NEXP; ++e) { p[e] = lp[e]; m = fmaxf(m, p[e]); }
    float s = 0.f;
    for (int e = 0; e < NEXP; ++e) { p[e] = expf(p[e] - m); s += p[e]; }
    float inv = 1.f / s;
    for (int e = 0; e < NEXP; ++e) p[e] *= inv;
    int ids[TOPK]; float gv[TOPK]; float gs = 0.f;
    for (int k = 0; k < TOPK; ++k) {
        int best = 0; float bv = -1.f;
        for (int e = 0; e < NEXP; ++e) if (p[e] > bv) { bv = p[e]; best = e; }
        ids[k] = best; gv[k] = bv; p[best] = -1.f; gs += bv;
    }
    float r = 1.f / (gs + 1e-9f);
    for (int k = 0; k < TOPK; ++k) {
        ti[t * TOPK + k] = ids[k];
        tg[t * TOPK + k] = gv[k] * r;
    }
}

// ---------------- 4) dispatch: 1 wave per expert, stable (t,k)-order slots --
__global__ __launch_bounds__(32) void k_dispatch(const int* __restrict__ ti,
                                                 int* __restrict__ etok,
                                                 int* __restrict__ ecnt,
                                                 int* __restrict__ slotv) {
    int e = blockIdx.x;
    int lane = threadIdx.x;
    int count = 0;
    for (int t0 = 0; t0 < NTOK; t0 += 32) {
        int t = t0 + lane;
        int kk = -1;
        #pragma unroll
        for (int k = 0; k < TOPK; ++k)
            if (ti[t * TOPK + k] == e) kk = k;       // experts distinct per token
        bool match = (kk >= 0);
        unsigned m32 = (unsigned)__ballot(match);    // wave32: low 32 bits
        int prefix = __popc(m32 & ((1u << lane) - 1u));
        if (match) {
            int slot = count + prefix;
            slotv[t * TOPK + kk] = slot;
            if (slot < CAP) etok[e * CAP + slot] = t;
        }
        count += __popc(m32);
    }
    if (lane == 0) ecnt[e] = count < CAP ? count : CAP;
}

// ---------------- WMMA GEMM tiles: 64x64 block, BK=32, 128 thr = 4 waves ----
// LDS rows padded to 80B -> conflict-free 16B fragment reads; double-buffered,
// filled by GLOBAL_LOAD_ASYNC_TO_LDS_B128 (ASYNCcnt), one barrier per k-step.
#define LDA 40                  // padded LDS row stride in bf16 elems
#define BUFE (64 * LDA)         // elems per LDS buffer

union Frag { bf16x16 v; uint4 q[2]; };

// gate+up fused with SiLU epilogue: Act[e, row, f] = silu(X Wg^T) * (X Wu^T)
__global__ __launch_bounds__(128) void k_gemm_gate_up(
        const u16* __restrict__ X, const u16* __restrict__ Wg,
        const u16* __restrict__ Wu, u16* __restrict__ Act,
        const int* __restrict__ etok, const int* __restrict__ ecnt,
        int rows, int Kdim) {
    __shared__ __align__(16) u16 sA[2 * BUFE], sG[2 * BUFE], sU[2 * BUFE];
    int e = blockIdx.z;
    int cnt = ecnt ? ecnt[e] : rows;
    int m0 = blockIdx.x * 64;
    if (m0 >= cnt) return;
    int f0 = blockIdx.y * 64;
    const u16* wg = Wg + (size_t)e * FDIM * Kdim;
    const u16* wu = Wu + (size_t)e * FDIM * Kdim;
    u16* act = Act + (size_t)e * rows * FDIM;

    int tid = threadIdx.x;
    int r = tid >> 1, part = tid & 1;
    int row = m0 + r;
    int tok = etok ? ((row < cnt) ? etok[e * CAP + row] : 0) : row;
    const u16* aSrc = X  + (size_t)tok * Kdim      + part * 16;
    const u16* gSrc = wg + (size_t)(f0 + r) * Kdim + part * 16;
    const u16* uSrc = wu + (size_t)(f0 + r) * Kdim + part * 16;
    int sOff = r * LDA + part * 16;
    u32 laA = lds_off(&sA[sOff]), laG = lds_off(&sG[sOff]), laU = lds_off(&sU[sOff]);

    int lane = tid & 31, wv = tid >> 5;
    int half = lane >> 4, l16 = lane & 15;
    int aLds = (wv * 16 + l16) * LDA;

    f32x8 z = {};
    f32x8 accg[4], accu[4];
    #pragma unroll
    for (int s = 0; s < 4; ++s) { accg[s] = z; accu[s] = z; }

    auto issue = [&](int k0, int sel) {
        u32 b = (u32)(sel * BUFE * 2);               // byte offset of buffer
        async_copy32(laA + b, aSrc + k0);
        async_copy32(laG + b, gSrc + k0);
        async_copy32(laU + b, uSrc + k0);
    };

    const int nIter = Kdim / 32;
    issue(0, 0);
    for (int it = 0; it < nIter; ++it) {
        wait_async0();                                // own buf[it&1] loads done
        __syncthreads();                              // all waves done reading other buf
        if (it + 1 < nIter) issue((it + 1) * 32, (it + 1) & 1);
        int so = (it & 1) * BUFE;
        // load ALL fragments first (distinct regs) so ds_loads overlap the
        // WMMA burst with partial dscnt waits instead of serializing
        Frag A, Bg[4], Bu[4];
        A.q[0] = *(const uint4*)&sA[so + aLds + half * 8];       // K 0-7 | 8-15
        A.q[1] = *(const uint4*)&sA[so + aLds + 16 + half * 8];  // K 16-23|24-31
        #pragma unroll
        for (int s = 0; s < 4; ++s) {
            int bO = so + (s * 16 + l16) * LDA + half * 8;       // K 0-7 | 8-15, 16-23|24-31
            Bg[s].q[0] = *(const uint4*)&sG[bO];
            Bg[s].q[1] = *(const uint4*)&sG[bO + 16];
            Bu[s].q[0] = *(const uint4*)&sU[bO];
            Bu[s].q[1] = *(const uint4*)&sU[bO + 16];
        }
        #pragma unroll
        for (int s = 0; s < 4; ++s) {
            accg[s] = __builtin_amdgcn_wmma_f32_16x16x32_bf16(
                false, A.v, false, Bg[s].v, (short)0, accg[s], false, false);
            accu[s] = __builtin_amdgcn_wmma_f32_16x16x32_bf16(
                false, A.v, false, Bu[s].v, (short)0, accu[s], false, false);
        }
    }
    #pragma unroll
    for (int s = 0; s < 4; ++s) {
        int col = f0 + s * 16 + l16;
        #pragma unroll
        for (int v = 0; v < 8; ++v) {
            int rr = m0 + wv * 16 + half * 8 + v;                // C: M=v+8*half
            float hg = accg[s][v], hu = accu[s][v];
            float sv = hg / (1.f + __expf(-hg)) * hu;            // silu(hg)*hu
            act[(size_t)rr * FDIM + col] = f2bf(sv);
        }
    }
}

// down-projection: Out[e, row, n] = A Wd^T  (stored bf16, gate applied later)
__global__ __launch_bounds__(128) void k_gemm_down(
        const u16* __restrict__ Ain, const u16* __restrict__ W,
        u16* __restrict__ Out, const int* __restrict__ ecnt,
        int rows, int Ndim, int Kdim) {
    __shared__ __align__(16) u16 sA[2 * BUFE], sB[2 * BUFE];
    int e = blockIdx.z;
    int cnt = ecnt ? ecnt[e] : rows;
    int m0 = blockIdx.x * 64;
    if (m0 >= cnt) return;
    int n0 = blockIdx.y * 64;
    const u16* Ae = Ain + (size_t)e * rows * Kdim;
    const u16* We = W   + (size_t)e * Ndim * Kdim;
    u16* Oe = Out + (size_t)e * rows * Ndim;

    int tid = threadIdx.x;
    int r = tid >> 1, part = tid & 1;
    const u16* aSrc = Ae + (size_t)(m0 + r) * Kdim + part * 16;
    const u16* bSrc = We + (size_t)(n0 + r) * Kdim + part * 16;
    int sOff = r * LDA + part * 16;
    u32 laA = lds_off(&sA[sOff]), laB = lds_off(&sB[sOff]);

    int lane = tid & 31, wv = tid >> 5;
    int half = lane >> 4, l16 = lane & 15;
    int aLds = (wv * 16 + l16) * LDA;

    f32x8 z = {};
    f32x8 acc[4];
    #pragma unroll
    for (int s = 0; s < 4; ++s) acc[s] = z;

    auto issue = [&](int k0, int sel) {
        u32 b = (u32)(sel * BUFE * 2);
        async_copy32(laA + b, aSrc + k0);
        async_copy32(laB + b, bSrc + k0);
    };

    const int nIter = Kdim / 32;
    issue(0, 0);
    for (int it = 0; it < nIter; ++it) {
        wait_async0();
        __syncthreads();
        if (it + 1 < nIter) issue((it + 1) * 32, (it + 1) & 1);
        int so = (it & 1) * BUFE;
        Frag A, B[4];
        A.q[0] = *(const uint4*)&sA[so + aLds + half * 8];
        A.q[1] = *(const uint4*)&sA[so + aLds + 16 + half * 8];
        #pragma unroll
        for (int s = 0; s < 4; ++s) {
            int bO = so + (s * 16 + l16) * LDA + half * 8;
            B[s].q[0] = *(const uint4*)&sB[bO];
            B[s].q[1] = *(const uint4*)&sB[bO + 16];
        }
        #pragma unroll
        for (int s = 0; s < 4; ++s)
            acc[s] = __builtin_amdgcn_wmma_f32_16x16x32_bf16(
                false, A.v, false, B[s].v, (short)0, acc[s], false, false);
    }
    #pragma unroll
    for (int s = 0; s < 4; ++s) {
        int col = n0 + s * 16 + l16;
        #pragma unroll
        for (int v = 0; v < 8; ++v) {
            int rr = m0 + wv * 16 + half * 8 + v;
            Oe[(size_t)rr * Ndim + col] = f2bf(acc[s][v]);
        }
    }
}

// ---------------- 7) combine: shared avg + gated expert gather --------------
__global__ __launch_bounds__(128) void k_combine(
        const u16* __restrict__ shout, const u16* __restrict__ eout,
        const int* __restrict__ ti, const float* __restrict__ tg,
        const int* __restrict__ slotv, float* __restrict__ out) {
    int t = blockIdx.x;
    int d0 = threadIdx.x * 4;                    // 128 thr * 4 = 512 = D
    const u16* s0 = shout + (size_t)t * DDIM + d0;
    const u16* s1 = shout + ((size_t)NTOK + t) * DDIM + d0;
    float acc[4];
    #pragma unroll
    for (int j = 0; j < 4; ++j) acc[j] = 0.5f * (bf2f(s0[j]) + bf2f(s1[j]));
    #pragma unroll
    for (int k = 0; k < TOPK; ++k) {
        int e  = ti[t * TOPK + k];
        int sl = slotv[t * TOPK + k];
        float g = tg[t * TOPK + k];
        if (sl < CAP) {                          // capacity-dropped tokens skip
            const u16* ep = eout + ((size_t)e * CAP + sl) * DDIM + d0;
            #pragma unroll
            for (int j = 0; j < 4; ++j) acc[j] += g * bf2f(ep[j]);
        }
    }
    #pragma unroll
    for (int j = 0; j < 4; ++j) out[(size_t)t * DDIM + d0 + j] = acc[j];
}

// ---------------- host side -------------------------------------------------
extern "C" void kernel_launch(void* const* d_in, const int* in_sizes, int n_in,
                              void* d_out, int out_size, void* d_ws, size_t ws_size,
                              hipStream_t stream) {
    const float* x    = (const float*)d_in[0];   // [4,2048,512]
    const float* rw   = (const float*)d_in[1];   // [64,512]
    const float* bias = (const float*)d_in[2];   // [64]
    const float* swg  = (const float*)d_in[3];   // [2,256,512]
    const float* swu  = (const float*)d_in[4];
    const float* swd  = (const float*)d_in[5];   // [2,512,256]
    const float* wg   = (const float*)d_in[6];   // [64,256,512]
    const float* wu   = (const float*)d_in[7];
    const float* wd   = (const float*)d_in[8];   // [64,512,256]
    float* out = (float*)d_out;

    size_t off = 0;
    auto alloc = [&](size_t bytes) -> void* {
        off = (off + 255) & ~(size_t)255;
        void* p = (char*)d_ws + off;
        off += bytes;
        return p;
    };
    u16*   xb    = (u16*)alloc((size_t)NTOK * DDIM * 2);
    u16*   wgb   = (u16*)alloc((size_t)NEXP * FDIM * DDIM * 2);
    u16*   wub   = (u16*)alloc((size_t)NEXP * FDIM * DDIM * 2);
    u16*   wdb   = (u16*)alloc((size_t)NEXP * DDIM * FDIM * 2);
    u16*   sgb   = (u16*)alloc((size_t)NSH * FDIM * DDIM * 2);
    u16*   sub_  = (u16*)alloc((size_t)NSH * FDIM * DDIM * 2);
    u16*   sdb   = (u16*)alloc((size_t)NSH * DDIM * FDIM * 2);
    float* logits= (float*)alloc((size_t)NTOK * NEXP * 4);
    int*   ti    = (int*)alloc((size_t)NTOK * TOPK * 4);
    float* tg    = (float*)alloc((size_t)NTOK * TOPK * 4);
    int*   slotv = (int*)alloc((size_t)NTOK * TOPK * 4);
    int*   ecnt  = (int*)alloc((size_t)NEXP * 4);
    int*   etok  = (int*)alloc((size_t)NEXP * CAP * 4);
    u16*   act   = (u16*)alloc((size_t)NEXP * CAP * FDIM * 2);
    u16*   actsh = (u16*)alloc((size_t)NSH * NTOK * FDIM * 2);
    u16*   eout  = (u16*)alloc((size_t)NEXP * CAP * DDIM * 2);
    u16*   shout = (u16*)alloc((size_t)NSH * NTOK * DDIM * 2);
    (void)ws_size; (void)in_sizes; (void)n_in; (void)out_size;

    auto cvt = [&](const float* s, u16* d, size_t n) {
        int n4 = (int)(n / 4);
        int blocks = (n4 + 255) / 256; if (blocks > 4096) blocks = 4096;
        k_cvt<<<blocks, 256, 0, stream>>>(s, d, n4);
    };
    cvt(x,   xb,   (size_t)NTOK * DDIM);
    cvt(wg,  wgb,  (size_t)NEXP * FDIM * DDIM);
    cvt(wu,  wub,  (size_t)NEXP * FDIM * DDIM);
    cvt(wd,  wdb,  (size_t)NEXP * DDIM * FDIM);
    cvt(swg, sgb,  (size_t)NSH * FDIM * DDIM);
    cvt(swu, sub_, (size_t)NSH * FDIM * DDIM);
    cvt(swd, sdb,  (size_t)NSH * DDIM * FDIM);

    k_router<<<NTOK, 64, 0, stream>>>(x, rw, bias, logits);
    k_topk<<<NTOK / 256, 256, 0, stream>>>(logits, ti, tg);
    k_dispatch<<<NEXP, 32, 0, stream>>>(ti, etok, ecnt, slotv);

    // shared experts: dense over all tokens (identity gather)
    k_gemm_gate_up<<<dim3(NTOK / 64, FDIM / 64, NSH), 128, 0, stream>>>(
        xb, sgb, sub_, actsh, nullptr, nullptr, NTOK, DDIM);
    // routed experts: gather rows through expert_tok, early-exit past count
    k_gemm_gate_up<<<dim3(CAP / 64, FDIM / 64, NEXP), 128, 0, stream>>>(
        xb, wgb, wub, act, etok, ecnt, CAP, DDIM);

    k_gemm_down<<<dim3(NTOK / 64, DDIM / 64, NSH), 128, 0, stream>>>(
        actsh, sdb, shout, nullptr, NTOK, DDIM, FDIM);
    k_gemm_down<<<dim3(CAP / 64, DDIM / 64, NEXP), 128, 0, stream>>>(
        act, wdb, eout, ecnt, CAP, DDIM, FDIM);

    k_combine<<<NTOK, 128, 0, stream>>>(shout, eout, ti, tg, slotv, out);
}
